// TkStateSpaceModule_22892175688154
// MI455X (gfx1250) — matrix operand order repaired and checked by reading, not compiled, not run on hardware
//
#include <hip/hip_runtime.h>

// ---------------------------------------------------------------------------
// S5-style diagonal SSM for MI455X (gfx1250, wave32, WMMA + TDM).
//   u = x @ B_bar^T   (GEMM1, f16 WMMA, f32 accum)
//   h_t = A_bar*h_{t-1} + u_t   (complex f32, chunked parallel scan)
//   y = Re(h)@Cr^T - Im(h)@Ci^T (GEMM2, f16 WMMA, f32 accum)
// 3 phases: local-scan (pass1) -> carry propagate -> fixup+output (pass2).
// x chunk staging uses the Tensor Data Mover (tensor_load_to_lds).
// ---------------------------------------------------------------------------

#define D_IN   128
#define D_ST   256
#define D_OUT  128
#define BATCH  8
#define SEQ    8192
#define CHUNK  64
#define NCHUNK (SEQ / CHUNK)   // 128

// LDS strides (elements), padded for bank-conflict relief but 16B aligned.
#define XS 136   // halves per x row   (128 + 8)
#define US 516   // floats per u row   (512 + 4)

// workspace byte offsets
#define OFF_ABAR 0                         // 256 float2            (2 KB)
#define OFF_BBAR 2048                      // 512x128 f16           (128 KB)
#define OFF_CMAT (2048 + 131072)           // 128x512 f16           (128 KB)
#define OFF_HEND (OFF_CMAT + 131072)       // 8*128*256 float2      (2 MB)
#define OFF_CARR (OFF_HEND + 2097152)      // 8*128*256 float2      (2 MB)

typedef __attribute__((ext_vector_type(16))) _Float16 v16h;
typedef __attribute__((ext_vector_type(8)))  _Float16 v8h;
typedef __attribute__((ext_vector_type(8)))  float    v8f;
typedef __attribute__((ext_vector_type(4)))  float    v4f;
typedef __attribute__((ext_vector_type(4)))  unsigned int u32x4;
typedef __attribute__((ext_vector_type(8)))  unsigned int u32x8;

// --- parameter prep: A_bar (complex f32) and B_bar (f16, [re;im] x 128) ----
__global__ void ssm_params(const float* __restrict__ llr,
                           const float* __restrict__ li,
                           const float* __restrict__ ldt,
                           const float* __restrict__ Bm,
                           float2* __restrict__ Abar,
                           _Float16* __restrict__ Bbar) {
  const int s = blockIdx.x;     // 0..255 state
  const int d = threadIdx.x;    // 0..127 input dim
  const float lre = -expf(llr[s]);
  const float lim = li[s];
  float dt = expf(ldt[s]);
  dt = fminf(fmaxf(dt, 0.005f), 0.1f);
  const float dr  = 1.0f - 0.5f * dt * lre;   // denom = 1 - 0.5*dt*Lambda
  const float di  = -0.5f * dt * lim;
  const float inv = 1.0f / (dr * dr + di * di);
  const float nr  = 1.0f + 0.5f * dt * lre;   // num = 1 + 0.5*dt*Lambda
  const float ni  = 0.5f * dt * lim;
  if (d == 0) {
    Abar[s] = make_float2((nr * dr + ni * di) * inv,
                          (ni * dr - nr * di) * inv);
  }
  const float sr = dt * dr * inv;             // dt/denom
  const float si = -dt * di * inv;
  const float bv = Bm[s * D_IN + d];
  Bbar[(size_t)s * D_IN + d]          = (_Float16)(sr * bv);
  Bbar[(size_t)(s + D_ST) * D_IN + d] = (_Float16)(si * bv);
}

// --- Cmat[o][k] f16 : k<256 -> Cr[o][k], k>=256 -> -Ci[o][k-256] -----------
__global__ void ssm_cmat(const float* __restrict__ Cr,
                         const float* __restrict__ Ci,
                         _Float16* __restrict__ Cmat) {
  const int o = blockIdx.x;     // 0..127
  const int k = threadIdx.x;    // 0..255
  Cmat[(size_t)o * 512 + k]       = (_Float16)(Cr[o * D_ST + k]);
  Cmat[(size_t)o * 512 + 256 + k] = (_Float16)(-Ci[o * D_ST + k]);
}

// --- sequential carry propagation across chunks ----------------------------
__global__ void ssm_carry(const float2* __restrict__ Abar,
                          const float2* __restrict__ hend,
                          float2* __restrict__ carry) {
  const int b = blockIdx.x;     // batch
  const int s = threadIdx.x;    // state
  const float2 A = Abar[s];
  float pr = A.x, pi = A.y;     // A^1
  #pragma unroll
  for (int i = 0; i < 6; ++i) { // -> A^64 == A^CHUNK
    const float nr = pr * pr - pi * pi;
    const float ni = 2.0f * pr * pi;
    pr = nr; pi = ni;
  }
  float cr = 0.0f, ci = 0.0f;
  for (int c = 0; c < NCHUNK; ++c) {
    const int idx = (b * NCHUNK + c) * D_ST + s;
    carry[idx] = make_float2(cr, ci);
    const float2 e = hend[idx];
    const float nr = pr * cr - pi * ci + e.x;
    const float ni = pr * ci + pi * cr + e.y;
    cr = nr; ci = ni;
  }
}

// --- main fused kernel: TDM stage -> GEMM1 -> scan (-> GEMM2 in pass 2) ----
template <int PASS>
__global__ __launch_bounds__(256) void ssm_main(
    const float* __restrict__ x,
    const _Float16* __restrict__ Bbar,
    const _Float16* __restrict__ Cmat,
    const float2* __restrict__ Abar,
    const float2* __restrict__ carry,
    float2* __restrict__ hend,
    float* __restrict__ y) {
  __shared__ _Float16 xlds[CHUNK * XS];   // ~17 KB
  __shared__ float    ulds[CHUNK * US];   // ~129 KB (x f32 staged here first)
  const int c     = blockIdx.x;
  const int b     = blockIdx.y;
  const int tid   = threadIdx.x;
  const int lane  = tid & 31;
  const int wave  = tid >> 5;
  const int m     = lane & 15;     // row/col within 16-tile
  const int lh    = lane >> 4;     // lane half
  const int tbase = c * CHUNK;

  // ---- TDM: DMA the contiguous 32KB x chunk into ulds (as raw f32) --------
  if (wave == 0) {
    const unsigned long long gaddr =
        (unsigned long long)(const void*)(x + ((size_t)b * SEQ + tbase) * D_IN);
    const unsigned lds_off = (unsigned)(unsigned long long)(const void*)&ulds[0];
    u32x4 g0;
    g0[0] = 1u;                                          // count=1, user mode
    g0[1] = lds_off;                                     // lds_addr
    g0[2] = (unsigned)(gaddr & 0xFFFFFFFFull);           // global_addr[31:0]
    g0[3] = ((unsigned)(gaddr >> 32) & 0x01FFFFFFu)      // global_addr[56:32]
            | 0x80000000u;                               // type=2 (image)
    u32x8 g1 = {};
    g1[0] = 0x00020000u;                                 // data_size=4B
    g1[1] = (unsigned)(CHUNK * D_IN) << 16;              // tensor_dim0 lo16
    g1[2] = (unsigned)(CHUNK * D_IN) >> 16;              // tensor_dim0 hi16
    g1[3] = (unsigned)(CHUNK * D_IN) << 16;              // tile_dim0 (1-D tile)
    g1[5] = (unsigned)(CHUNK * D_IN);                    // tensor_dim0_stride
    asm volatile("tensor_load_to_lds %0, %1" :: "s"(g0), "s"(g1) : "memory");
    __builtin_amdgcn_s_wait_tensorcnt(0);
  }
  __syncthreads();

  // ---- convert staged x f32 (in ulds) -> f16 into xlds --------------------
  {
    #pragma unroll
    for (int i = tid; i < CHUNK * D_IN / 4; i += 256) {
      const v4f v = *(const v4f*)&ulds[i * 4];
      const int t = (i * 4) / D_IN;
      const int d = (i * 4) % D_IN;
      _Float16* p = &xlds[t * XS + d];
      p[0] = (_Float16)v[0]; p[1] = (_Float16)v[1];
      p[2] = (_Float16)v[2]; p[3] = (_Float16)v[3];
    }
  }
  __syncthreads();

  const int mt    = wave & 3;   // row tile   (4 x 16 rows = CHUNK)
  const int chalf = wave >> 2;  // column half
  const int t0    = mt * 16;

  // ---- GEMM1: u[t][s(512)] = sum_d x[t][d] * Bbar[s][d] -> ulds -----------
  {
    // hoist the 4 A fragments (K = 0..127) for this wave's row tile
    v16h afr[4];
    #pragma unroll
    for (int kt = 0; kt < 4; ++kt) {
      const _Float16* prow = &xlds[(t0 + m) * XS + kt * 32 + lh * 16];
      const v8h lo = *(const v8h*)(prow);        // K = kb+lh*16   .. +7
      const v8h hi = *(const v8h*)(prow + 8);    // K = kb+lh*16+8 .. +7
      v16h a;
      #pragma unroll
      for (int j = 0; j < 8; ++j) { a[j] = lo[j]; a[8 + j] = hi[j]; }
      afr[kt] = a;
    }
    for (int nt = 0; nt < 16; ++nt) {
      const int s0 = (chalf * 16 + nt) * 16;
      v8f acc = {};
      #pragma unroll
      for (int kt = 0; kt < 4; ++kt) {
        const v16h bf = *(const v16h*)(Bbar + (size_t)(s0 + m) * D_IN +
                                       kt * 32 + lh * 16);
        acc = __builtin_amdgcn_wmma_f32_16x16x32_f16(
            false, afr[kt], false, bf, (short)0, acc, false, false);
      }
      #pragma unroll
      for (int r = 0; r < 8; ++r)
        ulds[(t0 + lh * 8 + r) * US + s0 + m] = acc[r];
    }
  }
  __syncthreads();

  // ---- complex diagonal scan: one state per thread ------------------------
  {
    const int s = tid;
    const float2 A = Abar[s];
    float hr, hi2;
    if (PASS == 2) {
      const float2 cv = carry[(b * NCHUNK + c) * D_ST + s];
      hr = cv.x; hi2 = cv.y;
    } else {
      hr = 0.0f; hi2 = 0.0f;
    }
    #pragma unroll 4
    for (int t = 0; t < CHUNK; ++t) {
      const float ur = ulds[t * US + s];
      const float ui = ulds[t * US + 256 + s];
      const float nr = fmaf(hr, A.x, fmaf(-hi2, A.y, ur));
      const float ni = fmaf(hr, A.y, fmaf(hi2, A.x, ui));
      hr = nr; hi2 = ni;
      if (PASS == 2) {                       // h overwrites u in place
        ulds[t * US + s]       = hr;
        ulds[t * US + 256 + s] = hi2;
      }
    }
    if (PASS == 1)
      hend[(b * NCHUNK + c) * D_ST + s] = make_float2(hr, hi2);
  }
  if (PASS == 1) return;
  __syncthreads();

  // ---- GEMM2: y[t][o] = sum_k h[t][k] * Cmat[o][k] ------------------------
  {
    v8f acc[4] = {};
    for (int kt = 0; kt < 16; ++kt) {
      const int kb = kt * 32;
      const float* prow = &ulds[(t0 + m) * US + kb + lh * 16];
      const v4f p0 = *(const v4f*)(prow);
      const v4f p1 = *(const v4f*)(prow + 4);
      const v4f p2 = *(const v4f*)(prow + 8);
      const v4f p3 = *(const v4f*)(prow + 12);
      v16h a;
      #pragma unroll
      for (int j = 0; j < 4; ++j) {
        a[j]      = (_Float16)p0[j];
        a[4 + j]  = (_Float16)p1[j];
        a[8 + j]  = (_Float16)p2[j];
        a[12 + j] = (_Float16)p3[j];
      }
      #pragma unroll
      for (int ot = 0; ot < 4; ++ot) {
        const int o0 = (chalf * 4 + ot) * 16;
        const v16h bf = *(const v16h*)(Cmat + (size_t)(o0 + m) * 512 +
                                       kb + lh * 16);
        acc[ot] = __builtin_amdgcn_wmma_f32_16x16x32_f16(
            false, a, false, bf, (short)0, acc[ot], false, false);
      }
    }
    #pragma unroll
    for (int ot = 0; ot < 4; ++ot) {
      const int o0 = (chalf * 4 + ot) * 16;
      float* yrow = y + ((size_t)b * SEQ + tbase + t0 + lh * 8) * D_OUT +
                    o0 + m;
      #pragma unroll
      for (int r = 0; r < 8; ++r)
        yrow[(size_t)r * D_OUT] = acc[ot][r];
    }
  }
}

extern "C" void kernel_launch(void* const* d_in, const int* in_sizes, int n_in,
                              void* d_out, int out_size, void* d_ws,
                              size_t ws_size, hipStream_t stream) {
  (void)in_sizes; (void)n_in; (void)out_size; (void)ws_size;
  const float* x   = (const float*)d_in[0];
  const float* llr = (const float*)d_in[1];
  const float* li  = (const float*)d_in[2];
  const float* ldt = (const float*)d_in[3];
  const float* Bm  = (const float*)d_in[4];
  const float* Cr  = (const float*)d_in[5];
  const float* Ci  = (const float*)d_in[6];
  float* y = (float*)d_out;

  char* ws = (char*)d_ws;
  float2*   Abar = (float2*)(ws + OFF_ABAR);
  _Float16* Bbar = (_Float16*)(ws + OFF_BBAR);
  _Float16* Cmat = (_Float16*)(ws + OFF_CMAT);
  float2*   hend = (float2*)(ws + OFF_HEND);
  float2*   carr = (float2*)(ws + OFF_CARR);

  ssm_params<<<dim3(D_ST), dim3(D_IN), 0, stream>>>(llr, li, ldt, Bm, Abar, Bbar);
  ssm_cmat<<<dim3(D_OUT), dim3(D_ST), 0, stream>>>(Cr, Ci, Cmat);
  ssm_main<1><<<dim3(NCHUNK, BATCH), dim3(256), 0, stream>>>(
      x, Bbar, Cmat, Abar, nullptr, hend, nullptr);
  ssm_carry<<<dim3(BATCH), dim3(D_ST), 0, stream>>>(Abar, hend, carr);
  ssm_main<2><<<dim3(NCHUNK, BATCH), dim3(256), 0, stream>>>(
      x, Bbar, Cmat, Abar, carr, hend, y);
}
